// CodeBook_31945966748125
// MI455X (gfx1250) — matrix operand order repaired and checked by reading, not compiled, not run on hardware
//
#include <hip/hip_runtime.h>
#include <hip/hip_bf16.h>
#include <float.h>

typedef __attribute__((ext_vector_type(16))) __bf16 v16bf;
typedef __attribute__((ext_vector_type(8)))  float  v8f;

#define NTOT 16384   // B*H*W
#define CIN  512
#define DLAT 256
#define KEMB 8192
#define HW   1024    // H*W

union Frag { v16bf v; uint4 q[2]; };

// ---------------------------------------------------------------------------
// Prep: w_proj f32 -> bf16 (natural [d][c] layout == WMMA B-frag layout)
// ---------------------------------------------------------------------------
__global__ void prep_w_kernel(const float* __restrict__ w, __bf16* __restrict__ wbf) {
    int i = blockIdx.x * 256 + threadIdx.x;
    if (i < DLAT * CIN) wbf[i] = (__bf16)w[i];
}

// ---------------------------------------------------------------------------
// Prep: emb f32 -> bf16 + per-entry squared norm (f32, exact)
// ---------------------------------------------------------------------------
__global__ void prep_emb_kernel(const float* __restrict__ emb,
                                __bf16* __restrict__ ebf,
                                float* __restrict__ embn) {
    __shared__ float s[256];
    int j = blockIdx.x, t = threadIdx.x;
    float v = emb[(size_t)j * DLAT + t];
    ebf[(size_t)j * DLAT + t] = (__bf16)v;
    s[t] = v * v;
    __syncthreads();
    for (int off = 128; off > 0; off >>= 1) {
        if (t < off) s[t] += s[t + off];
        __syncthreads();
    }
    if (t == 0) embn[j] = s[0];
}

// ---------------------------------------------------------------------------
// Projection GEMM: zp[n,d] = sum_c z[b,c,hw]*w[d,c] + bias[d]
// One block = 16 rows (one M tile), 8 waves x 2 N-tiles = all 256 d.
// bf16 WMMA 16x16x32, f32 accumulate; writes zp f32 + flat bf16.
// ---------------------------------------------------------------------------
__global__ __launch_bounds__(256) void proj_kernel(const float* __restrict__ z,
                                                   const __bf16* __restrict__ wbf,
                                                   const float* __restrict__ bias,
                                                   float* __restrict__ zp,
                                                   __bf16* __restrict__ flatb) {
    __shared__ __bf16 As[16 * 32];    // 1 KB
    __shared__ __bf16 Bs[256 * 32];   // 16 KB
    int t = threadIdx.x, lane = t & 31, w = t >> 5;
    int m0 = blockIdx.x * 16;
    int b = m0 >> 10, hw0 = m0 & (HW - 1);
    const float* zb = z + (size_t)b * CIN * HW + hw0;

    int nt0 = w * 2, nt1 = w * 2 + 1;
    v8f acc0 = {}, acc1 = {};
    int m = lane & 15, half = lane >> 4, n = m;

    for (int cc = 0; cc < CIN / 32; ++cc) {
        // A tile: 16 rows x 32 c, coalesced over m (contiguous hw)
        {
            int mm = t & 15, cl = t >> 4;                     // cl in 0..15
            As[mm * 32 + cl]      = (__bf16)zb[(size_t)(cc * 32 + cl) * HW + mm];
            As[mm * 32 + cl + 16] = (__bf16)zb[(size_t)(cc * 32 + cl + 16) * HW + mm];
        }
        // B tile: 256 d x 32 c, one row per thread (64B)
        {
            const uint4* src = (const uint4*)(wbf + (size_t)t * CIN + cc * 32);
            uint4* dst = (uint4*)(Bs + t * 32);
            dst[0] = src[0]; dst[1] = src[1]; dst[2] = src[2]; dst[3] = src[3];
        }
        __syncthreads();

        Frag a, b0, b1;
        const __bf16* ap = As + m * 32 + 8 * half;            // K = 8*half+0..7
        a.q[0] = *(const uint4*)(ap);
        a.q[1] = *(const uint4*)(ap + 16);                    // K = 16+8*half+0..7
        const __bf16* bp0 = Bs + (nt0 * 16 + n) * 32 + 8 * half;
        const __bf16* bp1 = Bs + (nt1 * 16 + n) * 32 + 8 * half;
        b0.q[0] = *(const uint4*)(bp0); b0.q[1] = *(const uint4*)(bp0 + 16);
        b1.q[0] = *(const uint4*)(bp1); b1.q[1] = *(const uint4*)(bp1 + 16);

        acc0 = __builtin_amdgcn_wmma_f32_16x16x32_bf16(false, a.v, false, b0.v,
                                                       (short)0, acc0, false, false);
        acc1 = __builtin_amdgcn_wmma_f32_16x16x32_bf16(false, a.v, false, b1.v,
                                                       (short)0, acc1, false, false);
        __syncthreads();
    }

    int d0 = nt0 * 16 + n, d1 = nt1 * 16 + n;
    float bi0 = bias[d0], bi1 = bias[d1];
#pragma unroll
    for (int r = 0; r < 8; ++r) {
        int row = m0 + r + 8 * half;
        float v0 = acc0[r] + bi0, v1 = acc1[r] + bi1;
        zp[(size_t)row * DLAT + d0] = v0;
        zp[(size_t)row * DLAT + d1] = v1;
        flatb[(size_t)row * DLAT + d0] = (__bf16)v0;
        flatb[(size_t)row * DLAT + d1] = (__bf16)v1;
    }
}

// ---------------------------------------------------------------------------
// Distance GEMM + fused argmin. Block = 64 rows x all K entries.
// Per wave: fixed 16-row tile, 2 independent 16-entry WMMA chains.
// A fragments live in registers for the whole K loop.
// emb tiles are DMA'd memory->LDS with GLOBAL_LOAD_ASYNC_TO_LDS_B128
// (ASYNCcnt) and double-buffered so the next tile streams in while the
// current one feeds the matrix pipes.
// d2' = ||emb_k||^2 - 2*dot (row-constant term dropped for argmin).
// ---------------------------------------------------------------------------
__global__ __launch_bounds__(256) void dist_kernel(const __bf16* __restrict__ flatb,
                                                   const __bf16* __restrict__ ebf,
                                                   const float* __restrict__ embn,
                                                   int* __restrict__ idx_out) {
    __shared__ __bf16 Bs[2][64 * 256];     // 2 x 32KB emb tiles (double buffer)
    __shared__ float  red_v[8][32][16];    // 16 KB
    __shared__ int    red_i[8][32][16];    // 16 KB
    int t = threadIdx.x, lane = t & 31, w = t >> 5;
    int m0 = blockIdx.x * 64;
    int mt = w >> 1, ntp = (w & 1) * 2;
    int m = lane & 15, half = lane >> 4, n = m;

    // Preload 8 A fragments (full D=256) straight from global; reused 128x.
    Frag a[8];
    const __bf16* arow = flatb + (size_t)(m0 + mt * 16 + m) * DLAT + 8 * half;
#pragma unroll
    for (int kc = 0; kc < 8; ++kc) {
        a[kc].q[0] = *(const uint4*)(arow + kc * 32);
        a[kc].q[1] = *(const uint4*)(arow + kc * 32 + 16);
    }

    // Per-thread async-DMA slice: 128 contiguous bytes of the 32KB tile.
    // The same IOFFSET is applied to both the LDS and global addresses
    // (ISA 08_async_tensor 4.4), so one address pair covers 8 x b128.
    unsigned lds0 = (unsigned)(uintptr_t)(&Bs[0][0]) + (unsigned)t * 128u;
    unsigned lds1 = lds0 + 32768u;
    unsigned long long g =
        (unsigned long long)(uintptr_t)ebf + (unsigned long long)t * 128ull;

#define ASYNC_TILE(ldsaddr, gaddr)                                             \
    asm volatile("global_load_async_to_lds_b128 %0, %1, off\n\t"               \
                 "global_load_async_to_lds_b128 %0, %1, off offset:16\n\t"     \
                 "global_load_async_to_lds_b128 %0, %1, off offset:32\n\t"     \
                 "global_load_async_to_lds_b128 %0, %1, off offset:48\n\t"     \
                 "global_load_async_to_lds_b128 %0, %1, off offset:64\n\t"     \
                 "global_load_async_to_lds_b128 %0, %1, off offset:80\n\t"     \
                 "global_load_async_to_lds_b128 %0, %1, off offset:96\n\t"     \
                 "global_load_async_to_lds_b128 %0, %1, off offset:112"        \
                 :: "v"(ldsaddr), "v"(gaddr) : "memory")

    ASYNC_TILE(lds0, g);            // chunk 0 -> buffer 0
    g += 32768ull;

    float bestv[2][8];
    int   besti[2][8];
#pragma unroll
    for (int c = 0; c < 2; ++c)
#pragma unroll
        for (int r = 0; r < 8; ++r) { bestv[c][r] = FLT_MAX; besti[c][r] = 0; }

    for (int jc = 0; jc < KEMB / 64; ++jc) {
        int p = jc & 1;
        // Prefetch next chunk into the other buffer (safe: previous
        // iteration's trailing barrier means nobody still reads it).
        if (jc + 1 < KEMB / 64) {
            unsigned ldsn = p ? lds0 : lds1;
            ASYNC_TILE(ldsn, g);
            g += 32768ull;
            // <=8 outstanding: current chunk's 8 loads complete (in-order),
            // only the just-issued prefetch remains in flight.
            asm volatile("s_wait_asynccnt 0x8" ::: "memory");
        } else {
            asm volatile("s_wait_asynccnt 0x0" ::: "memory");
        }
        __syncthreads();            // all waves' DMA slices visible

        const __bf16* B = &Bs[p][0];
        v8f acc0 = {}, acc1 = {};
#pragma unroll
        for (int kc = 0; kc < 8; ++kc) {
            Frag b0, b1;
            const __bf16* bp0 = B + ((ntp)     * 16 + n) * DLAT + kc * 32 + 8 * half;
            const __bf16* bp1 = B + ((ntp + 1) * 16 + n) * DLAT + kc * 32 + 8 * half;
            b0.q[0] = *(const uint4*)(bp0); b0.q[1] = *(const uint4*)(bp0 + 16);
            b1.q[0] = *(const uint4*)(bp1); b1.q[1] = *(const uint4*)(bp1 + 16);
            acc0 = __builtin_amdgcn_wmma_f32_16x16x32_bf16(false, a[kc].v, false, b0.v,
                                                           (short)0, acc0, false, false);
            acc1 = __builtin_amdgcn_wmma_f32_16x16x32_bf16(false, a[kc].v, false, b1.v,
                                                           (short)0, acc1, false, false);
        }

        int j0 = jc * 64;
        int j0n = j0 + ntp * 16 + n;
        float en0 = embn[j0n];
        float en1 = embn[j0n + 16];
#pragma unroll
        for (int r = 0; r < 8; ++r) {
            float d0 = en0 - 2.0f * acc0[r];
            float d1 = en1 - 2.0f * acc1[r];
            if (d0 < bestv[0][r]) { bestv[0][r] = d0; besti[0][r] = j0n; }
            if (d1 < bestv[1][r]) { bestv[1][r] = d1; besti[1][r] = j0n + 16; }
        }
        __syncthreads();            // done reading Bs[p]; next iter refills it
    }
#undef ASYNC_TILE

#pragma unroll
    for (int c = 0; c < 2; ++c)
#pragma unroll
        for (int r = 0; r < 8; ++r) {
            red_v[w][lane][c * 8 + r] = bestv[c][r];
            red_i[w][lane][c * 8 + r] = besti[c][r];
        }
    __syncthreads();

    if (t < 64) {
        int mm = t, mtt = mm >> 4, mrow = mm & 15, hh = mrow >> 3, rr = mrow & 7;
        float bv = FLT_MAX;
        int bi = 0x7fffffff;
        for (int wv = mtt * 2; wv <= mtt * 2 + 1; ++wv)
            for (int c = 0; c < 2; ++c)
                for (int nn = 0; nn < 16; ++nn) {
                    float v = red_v[wv][hh * 16 + nn][c * 8 + rr];
                    int   i = red_i[wv][hh * 16 + nn][c * 8 + rr];
                    if (v < bv || (v == bv && i < bi)) { bv = v; bi = i; }
                }
        idx_out[m0 + mm] = bi;
    }
}

// ---------------------------------------------------------------------------
// Output: out[b,d,hw] = zp + (emb[idx] - zp) (straight-through, f32 like ref),
// plus float(idx) appended. Block = (b,d), threads along hw (coalesced store).
// ---------------------------------------------------------------------------
__global__ __launch_bounds__(256) void out_kernel(const float* __restrict__ zp,
                                                  const float* __restrict__ emb,
                                                  const int* __restrict__ idx,
                                                  float* __restrict__ out,
                                                  float* __restrict__ out_idx) {
    int bd = blockIdx.x;              // b*256 + d
    int b = bd >> 8, d = bd & 255;
    int t = threadIdx.x;
#pragma unroll
    for (int k = 0; k < 4; ++k) {
        int hw = t + k * 256;
        int nrow = b * HW + hw;
        int id = idx[nrow];
        float zq  = emb[(size_t)id * DLAT + d];
        float zpv = zp[(size_t)nrow * DLAT + d];
        out[(size_t)bd * HW + hw] = zpv + (zq - zpv);
        if (d == 0) out_idx[nrow] = (float)id;
    }
}

// ---------------------------------------------------------------------------
extern "C" void kernel_launch(void* const* d_in, const int* in_sizes, int n_in,
                              void* d_out, int out_size, void* d_ws, size_t ws_size,
                              hipStream_t stream) {
    const float* z      = (const float*)d_in[0];   // [16,512,32,32]
    const float* w_proj = (const float*)d_in[1];   // [256,512]
    const float* b_proj = (const float*)d_in[2];   // [256]
    const float* emb    = (const float*)d_in[3];   // [8192,256]

    uint8_t* p = (uint8_t*)d_ws;
    float*  zp    = (float*)p;   p += (size_t)NTOT * DLAT * sizeof(float);   // 16 MB
    __bf16* flatb = (__bf16*)p;  p += (size_t)NTOT * DLAT * sizeof(__bf16);  //  8 MB
    __bf16* ebf   = (__bf16*)p;  p += (size_t)KEMB * DLAT * sizeof(__bf16);  //  4 MB
    float*  embn  = (float*)p;   p += (size_t)KEMB * sizeof(float);
    __bf16* wbf   = (__bf16*)p;  p += (size_t)DLAT * CIN * sizeof(__bf16);
    int*    idxw  = (int*)p;

    float* out     = (float*)d_out;
    float* out_idx = out + (size_t)NTOT * DLAT;    // 4194304 .. 4210688

    prep_w_kernel  <<<(DLAT * CIN + 255) / 256, 256, 0, stream>>>(w_proj, wbf);
    prep_emb_kernel<<<KEMB, 256, 0, stream>>>(emb, ebf, embn);
    proj_kernel    <<<NTOT / 16, 256, 0, stream>>>(z, wbf, b_proj, zp, flatb);
    dist_kernel    <<<NTOT / 64, 256, 0, stream>>>(flatb, ebf, embn, idxw);
    out_kernel     <<<16 * DLAT, 256, 0, stream>>>(zp, emb, idxw, out, out_idx);
}
